// HSTULayer_34471407518414
// MI455X (gfx1250) — matrix operand order repaired and checked by reading, not compiled, not run on hardware
//
#include <hip/hip_runtime.h>
#include <hip/hip_bf16.h>

// ---------------------------------------------------------------------------
// HSTU layer for gfx1250 (MI455X).  All heavy matmuls use
// v_wmma_f32_16x16x32_bf16 (bf16 operands, fp32 accumulate).
// B=2, T=1024, D=1024, H=16, hd=64.
// ---------------------------------------------------------------------------

typedef __bf16 bf16;
typedef __attribute__((ext_vector_type(8)))  bf16  v8bf;
typedef __attribute__((ext_vector_type(16))) bf16  v16bf;
typedef __attribute__((ext_vector_type(8)))  float v8f;

#define HB   2
#define HT   1024
#define HD_  1024
#define NH   16
#define HDH  64      // head dim

__device__ __forceinline__ v8f wmma_bf16(v16bf a, v16bf b, v8f c) {
  return __builtin_amdgcn_wmma_f32_16x16x32_bf16(false, a, false, b, (short)0, c,
                                                 false, false);
}

// A-matrix fragment, 16x32 bf16 (M x K), source row-major [row][k], stride ld.
// Per ISA 7.12.2: lane m = lane%16, half = lane/16;
// elems 0..7  <-> K = k0 + half*8 + (0..7)
// elems 8..15 <-> K = k0 + 16 + half*8 + (0..7)
__device__ __forceinline__ v16bf load_afrag(const bf16* base, int row0, int k0,
                                            int ld, int lane) {
  const int m = lane & 15, half = lane >> 4;
  const bf16* p = base + (size_t)(row0 + m) * ld + k0 + half * 8;
  v8bf lo = *(const v8bf*)p;
  v8bf hi = *(const v8bf*)(p + 16);
  v16bf a;
#pragma unroll
  for (int i = 0; i < 8; ++i) { a[i] = lo[i]; a[i + 8] = hi[i]; }
  return a;
}

// B-matrix fragment, 32x16 bf16 (K x N) where B[k][n] = Src[n0+n][c0+k]
// (K runs along the contiguous axis of Src).  Per ISA: lane n = lane%16,
// elems 0..7 <-> K = half*8 + (0..7), elems 8..15 <-> K = 16 + half*8 + (0..7).
__device__ __forceinline__ v16bf load_bfrag(const bf16* base, int n0, int c0,
                                            int ld, int lane) {
  const int n = lane & 15, half = lane >> 4;
  const bf16* p = base + (size_t)(n0 + n) * ld + c0 + half * 8;
  v8bf lo = *(const v8bf*)p;
  v8bf hi = *(const v8bf*)(p + 16);
  v16bf b;
#pragma unroll
  for (int i = 0; i < 8; ++i) { b[i] = lo[i]; b[i + 8] = hi[i]; }
  return b;
}

__device__ __forceinline__ float silu_f(float c) {
  return c / (1.0f + __expf(-c));
}

// ---------------------------------------------------------------------------
// fp32 -> bf16 conversion
// ---------------------------------------------------------------------------
__global__ __launch_bounds__(256) void cvt_kernel(const float* __restrict__ src,
                                                  bf16* __restrict__ dst, int n) {
  int i = blockIdx.x * 256 + threadIdx.x;
  if (i < n) dst[i] = (bf16)src[i];
}

// ---------------------------------------------------------------------------
// GEMM1: h = silu(X @ W_in^T + b_in), routed into U(fp32), V/Q/K (bf16).
// X: [2048 x 1024] bf16, W: [4096 x 1024] bf16.  Block = 128 thr = 4 waves,
// block tile 64(M) x 128(N), wave tile 32 x 64.
// ---------------------------------------------------------------------------
__global__ __launch_bounds__(128) void gemm_in_kernel(
    const bf16* __restrict__ X, const bf16* __restrict__ W,
    const float* __restrict__ bin, float* __restrict__ U,
    bf16* __restrict__ V, bf16* __restrict__ Q, bf16* __restrict__ Kc) {
  const int lane = threadIdx.x & 31;
  const int wave = threadIdx.x >> 5;
  const int m0 = blockIdx.x * 64 + (wave & 1) * 32;
  const int n0 = blockIdx.y * 128 + (wave >> 1) * 64;
  const int hi = lane >> 4, lo = lane & 15;

  v8f acc[2][4] = {};
  for (int k0 = 0; k0 < HD_; k0 += 32) {
    v16bf a0 = load_afrag(X, m0, k0, HD_, lane);
    v16bf a1 = load_afrag(X, m0 + 16, k0, HD_, lane);
#pragma unroll
    for (int nt = 0; nt < 4; ++nt) {
      v16bf bfr = load_bfrag(W, n0 + nt * 16, k0, HD_, lane);
      acc[0][nt] = wmma_bf16(a0, bfr, acc[0][nt]);
      acc[1][nt] = wmma_bf16(a1, bfr, acc[1][nt]);
    }
  }
  const int seg = n0 >> 10;  // 0:U 1:V 2:Q 3:K  (N-block never straddles)
#pragma unroll
  for (int mt = 0; mt < 2; ++mt)
#pragma unroll
    for (int nt = 0; nt < 4; ++nt)
#pragma unroll
      for (int r = 0; r < 8; ++r) {
        const int row = m0 + mt * 16 + hi * 8 + r;
        const int ng = n0 + nt * 16 + lo;
        const float sv = silu_f(acc[mt][nt][r] + bin[ng]);
        const size_t idx = (size_t)row * HD_ + (ng & 1023);
        if (seg == 0)      U[idx] = sv;
        else if (seg == 1) V[idx] = (bf16)sv;
        else if (seg == 2) Q[idx] = (bf16)sv;
        else               Kc[idx] = (bf16)sv;
      }
}

// ---------------------------------------------------------------------------
// Attention: per wave, one 16-query strip of one (b,h); causal loop over
// 32-key steps.  QK^T and S*V via WMMA; S and V^T staged through LDS.
// attention = silu(qk*alpha + rab + mask) / denom ;  denom factored per-row.
// ---------------------------------------------------------------------------
__global__ __launch_bounds__(128) void attn_kernel(
    const bf16* __restrict__ Q, const bf16* __restrict__ Kc,
    const bf16* __restrict__ V, const unsigned char* __restrict__ kpm,
    const float* __restrict__ rab_emb, float* __restrict__ AV) {
  const int lane = threadIdx.x & 31;
  const int wave = threadIdx.x >> 5;
  const int b = blockIdx.x >> 8;         // grid.x = B*H*16 = 512
  const int h = (blockIdx.x >> 4) & 15;
  const int g = blockIdx.x & 15;
  const int t0 = (g * 4 + wave) * 16;
  const int hi = lane >> 4, lo = lane & 15;
  const float alpha = 0.125f;            // hd^-0.5, hd=64

  __shared__ float rb[HT];               // rel-pos bias per delta, this head
  __shared__ float dinv[64];             // 1/denom for the block's 64 rows
  __shared__ bf16 sS[4][16 * 32];        // per-wave S tile (16 x 32)
  __shared__ bf16 sVT[4][HDH * 32];      // per-wave V^T tile (64 x 32)

  // T5-style log-bucket relative position bias table (delta = t - s >= 0)
  for (int d = threadIdx.x; d < HT; d += 128) {
    int bucket;
    if (d < 16) {
      bucket = d;
    } else {
      // 16 + int( log(d/16 + 1e-6)/log(8) * 16 ), clamped to 31
      int lb = 16 + (int)(logf((float)d * (1.0f / 16.0f) + 1e-6f) *
                          (16.0f / logf(8.0f)));
      bucket = lb < 31 ? lb : 31;
    }
    rb[d] = rab_emb[bucket * NH + h];
  }
  // denom[t] = max(1, #{ s <= t : key s valid })
  if (threadIdx.x < 64) {
    const int t = g * 64 + threadIdx.x;
    int cnt = 0;
    for (int s = 0; s <= t; ++s) cnt += (kpm[b * HT + s] == 0) ? 1 : 0;
    dinv[threadIdx.x] = 1.0f / (float)(cnt < 1 ? 1 : cnt);
  }
  __syncthreads();

  const bf16* Qp = Q + (size_t)b * HT * HD_ + h * HDH;
  const bf16* Kp = Kc + (size_t)b * HT * HD_ + h * HDH;
  const bf16* Vp = V + (size_t)b * HT * HD_ + h * HDH;
  bf16* S = sS[wave];
  bf16* VT = sVT[wave];

  v16bf qa[2];
  qa[0] = load_afrag(Qp, t0, 0, HD_, lane);
  qa[1] = load_afrag(Qp, t0, 32, HD_, lane);
  v8f acc[4] = {};

  for (int s0 = 0; s0 < t0 + 16; s0 += 32) {
    // stage V^T[d][s_local] (transpose so B-fragments read contiguously)
    {
      const bf16* vr = Vp + (size_t)(s0 + lane) * HD_;
#pragma unroll
      for (int d = 0; d < HDH; ++d) VT[d * 32 + lane] = vr[d];
    }
    // two 16x16 S tiles covering s_local 0..31
#pragma unroll
    for (int st = 0; st < 2; ++st) {
      v8f sc = {};
      v16bf kb0 = load_bfrag(Kp, s0 + st * 16, 0, HD_, lane);
      sc = wmma_bf16(qa[0], kb0, sc);
      v16bf kb1 = load_bfrag(Kp, s0 + st * 16, 32, HD_, lane);
      sc = wmma_bf16(qa[1], kb1, sc);
      const int s = s0 + st * 16 + lo;                  // this lane's key col
      const int pv = (kpm[b * HT + s] == 0) ? 1 : 0;    // key not padded
#pragma unroll
      for (int r = 0; r < 8; ++r) {
        const int t = t0 + hi * 8 + r;
        float val = 0.0f;
        if (pv && s <= t) val = silu_f(sc[r] * alpha + rb[t - s]);
        S[(hi * 8 + r) * 32 + st * 16 + lo] = (bf16)val;
      }
    }
    asm volatile("s_wait_dscnt 0" ::: "memory");  // LDS RAW: S/VT -> frags
    v16bf sa = load_afrag(S, 0, 0, 32, lane);
#pragma unroll
    for (int nt = 0; nt < 4; ++nt) {
      v16bf vb = load_bfrag(VT, nt * 16, 0, 32, lane);
      acc[nt] = wmma_bf16(sa, vb, acc[nt]);
    }
  }

  float* outp = AV + (size_t)b * HT * HD_ + h * HDH;
#pragma unroll
  for (int nt = 0; nt < 4; ++nt)
#pragma unroll
    for (int r = 0; r < 8; ++r) {
      const int tl = hi * 8 + r;
      outp[(size_t)(t0 + tl) * HD_ + nt * 16 + lo] =
          acc[nt][r] * dinv[wave * 16 + tl];
    }
}

// ---------------------------------------------------------------------------
// LayerNorm(AV) * gamma + beta, then * U, cast bf16 -> NU.  One block / row.
// ---------------------------------------------------------------------------
__global__ __launch_bounds__(256) void ln_kernel(
    const float* __restrict__ AV, const float* __restrict__ U,
    const float* __restrict__ gamma, const float* __restrict__ beta,
    bf16* __restrict__ NU) {
  const int row = blockIdx.x;
  const int lane = threadIdx.x & 31;
  const int wave = threadIdx.x >> 5;
  __shared__ float rs[8], rs2[8];
  __shared__ float smu, srstd;
  const float* avr = AV + (size_t)row * HD_;
  float s = 0.0f, s2 = 0.0f;
  for (int i = threadIdx.x; i < HD_; i += 256) {
    float v = avr[i];
    s += v;
    s2 += v * v;
  }
#pragma unroll
  for (int o = 16; o > 0; o >>= 1) {
    s += __shfl_xor(s, o);
    s2 += __shfl_xor(s2, o);
  }
  if (lane == 0) { rs[wave] = s; rs2[wave] = s2; }
  __syncthreads();
  if (threadIdx.x == 0) {
    float ts = 0.0f, ts2 = 0.0f;
#pragma unroll
    for (int w = 0; w < 8; ++w) { ts += rs[w]; ts2 += rs2[w]; }
    const float mu = ts * (1.0f / HD_);
    smu = mu;
    srstd = rsqrtf(ts2 * (1.0f / HD_) - mu * mu + 1e-5f);
  }
  __syncthreads();
  const float mu = smu, rstd = srstd;
  const float* ur = U + (size_t)row * HD_;
  bf16* nr = NU + (size_t)row * HD_;
  for (int i = threadIdx.x; i < HD_; i += 256) {
    const float nv = (avr[i] - mu) * rstd * gamma[i] + beta[i];
    nr[i] = (bf16)(nv * ur[i]);
  }
}

// ---------------------------------------------------------------------------
// GEMM2: out = x + NU @ W_out^T + b_out.  Same tiling as GEMM1.
// ---------------------------------------------------------------------------
__global__ __launch_bounds__(128) void gemm_out_kernel(
    const bf16* __restrict__ NU, const bf16* __restrict__ W,
    const float* __restrict__ bout, const float* __restrict__ x,
    float* __restrict__ out) {
  const int lane = threadIdx.x & 31;
  const int wave = threadIdx.x >> 5;
  const int m0 = blockIdx.x * 64 + (wave & 1) * 32;
  const int n0 = blockIdx.y * 128 + (wave >> 1) * 64;
  const int hi = lane >> 4, lo = lane & 15;

  v8f acc[2][4] = {};
  for (int k0 = 0; k0 < HD_; k0 += 32) {
    v16bf a0 = load_afrag(NU, m0, k0, HD_, lane);
    v16bf a1 = load_afrag(NU, m0 + 16, k0, HD_, lane);
#pragma unroll
    for (int nt = 0; nt < 4; ++nt) {
      v16bf bfr = load_bfrag(W, n0 + nt * 16, k0, HD_, lane);
      acc[0][nt] = wmma_bf16(a0, bfr, acc[0][nt]);
      acc[1][nt] = wmma_bf16(a1, bfr, acc[1][nt]);
    }
  }
#pragma unroll
  for (int mt = 0; mt < 2; ++mt)
#pragma unroll
    for (int nt = 0; nt < 4; ++nt)
#pragma unroll
      for (int r = 0; r < 8; ++r) {
        const int row = m0 + mt * 16 + hi * 8 + r;
        const int n = n0 + nt * 16 + lo;
        const size_t idx = (size_t)row * HD_ + n;
        out[idx] = x[idx] + acc[mt][nt][r] + bout[n];
      }
}

// ---------------------------------------------------------------------------
// launch
// ---------------------------------------------------------------------------
extern "C" void kernel_launch(void* const* d_in, const int* in_sizes, int n_in,
                              void* d_out, int out_size, void* d_ws,
                              size_t ws_size, hipStream_t stream) {
  (void)in_sizes; (void)n_in; (void)out_size; (void)ws_size;
  const float* x = (const float*)d_in[0];
  // d_in[1] = attention_mask (pure causal; implemented analytically)
  const unsigned char* kpm = (const unsigned char*)d_in[2];
  const float* W_in = (const float*)d_in[3];
  const float* b_in = (const float*)d_in[4];
  const float* W_out = (const float*)d_in[5];
  const float* b_out = (const float*)d_in[6];
  const float* gamma = (const float*)d_in[7];
  const float* beta = (const float*)d_in[8];
  const float* rab_emb = (const float*)d_in[9];
  float* out = (float*)d_out;

  char* ws = (char*)d_ws;
  const size_t MB = (size_t)1 << 20;
  bf16* Xbf = (bf16*)(ws + 0 * MB);    // 2048x1024 bf16  (4 MB)
  bf16* Wi  = (bf16*)(ws + 4 * MB);    // 4096x1024 bf16  (8 MB)
  bf16* Wo  = (bf16*)(ws + 12 * MB);   // 1024x1024 bf16  (2 MB)
  float* U  = (float*)(ws + 14 * MB);  // 2048x1024 f32   (8 MB)
  bf16* Vb  = (bf16*)(ws + 22 * MB);   // 2048x1024 bf16  (4 MB)
  bf16* Qb  = (bf16*)(ws + 26 * MB);   // 2048x1024 bf16  (4 MB)
  bf16* Kb  = (bf16*)(ws + 30 * MB);   // 2048x1024 bf16  (4 MB)
  float* AV = (float*)(ws + 34 * MB);  // 2048x1024 f32   (8 MB)
  bf16* NU  = (bf16*)(ws + 42 * MB);   // 2048x1024 bf16  (4 MB)

  const int nX = HB * HT * HD_;        // 2 M
  const int nWi = 4 * HD_ * HD_;       // 4 M
  const int nWo = HD_ * HD_;           // 1 M
  cvt_kernel<<<(nX + 255) / 256, 256, 0, stream>>>(x, Xbf, nX);
  cvt_kernel<<<(nWi + 255) / 256, 256, 0, stream>>>(W_in, Wi, nWi);
  cvt_kernel<<<(nWo + 255) / 256, 256, 0, stream>>>(W_out, Wo, nWo);

  gemm_in_kernel<<<dim3(32, 32), 128, 0, stream>>>(Xbf, Wi, b_in, U, Vb, Qb, Kb);
  attn_kernel<<<HB * NH * 16, 128, 0, stream>>>(Qb, Kb, Vb, kpm, rab_emb, AV);
  ln_kernel<<<HB * HT, 256, 0, stream>>>(AV, U, gamma, beta, NU);
  gemm_out_kernel<<<dim3(32, 8), 128, 0, stream>>>(NU, Wo, b_out, x, out);
}
